// GNNTransformerModel_18305150615574
// MI455X (gfx1250) — compile-verified, matrix operand order checked by reading, not run on hardware
//
#include <hip/hip_runtime.h>
#include <stdint.h>

// ---------------------------------------------------------------------------
// Types for CDNA5 WMMA
// ---------------------------------------------------------------------------
typedef __bf16 bf16;
typedef __attribute__((ext_vector_type(16))) __bf16 v16bf;
typedef __attribute__((ext_vector_type(8)))  __bf16 v8bf;
typedef __attribute__((ext_vector_type(8)))  float  v8f;

#define N_NODES 50000
#define N_EDGES 1600000
#define SEQ     8
#define DIM     256
#define HIDN    256
#define HEADS   4
#define LAYERS  2
#define NCLS    10

// Stage B panel via CDNA5 async memory->LDS DMA (ASYNCcnt) instead of
// synchronous global_load + ds_store. Flip to 0 if assembler rejects it.
#ifndef ASYNC_STAGE
#define ASYNC_STAGE 1
#endif

__device__ __forceinline__ bf16 f2bf(float f) {
  union { float f; uint32_t u; } x; x.f = f;
  uint32_t r = x.u + 0x7FFFu + ((x.u >> 16) & 1u);   // round-to-nearest-even
  union { uint16_t u; bf16 b; } y; y.u = (uint16_t)(r >> 16);
  return y.b;
}
__device__ __forceinline__ float bf2f(bf16 b) {
  union { uint16_t u; bf16 b; } y; y.b = b;
  union { float f; uint32_t u; } x; x.u = ((uint32_t)y.u) << 16;
  return x.f;
}

union FragAB { v16bf v; v8bf h[2]; };

// ---------------------------------------------------------------------------
// Tiled WMMA GEMM:  C[M,Nc] = A[M,K](bf16,row-major) * B[Nc,K](bf16,row-major)^T
//                   + bias[Nc], optional ReLU; f32 and/or bf16 outputs.
// Block: 256 threads = 8 waves. Block tile: M=256 (2 x 16 rows per wave),
// N=64 (4 wmma tiles), K stepped by 32 with v_wmma_f32_16x16x32_bf16.
// B panel [64 x K] staged in LDS (async DMA on CDNA5), reused by both M tiles:
// per K-step/wave: 8 WMMA : 4 global_b128 : 8 ds_b128.
// Requires: K <= 256, K % 32 == 0, Nc % 64 == 0. (All call sites use K=256.)
// ---------------------------------------------------------------------------
#define BSTRIDE (256 + 8)   // bf16 elements; +16B pad avoids LDS bank conflicts

__global__ void __launch_bounds__(256)
k_gemm(const bf16* __restrict__ A, const bf16* __restrict__ B,
       const float* __restrict__ bias, float* __restrict__ Cf,
       bf16* __restrict__ Cb, int M, int Nc, int K, int relu)
{
  __shared__ bf16 Bs[64 * BSTRIDE];

  const int tid  = threadIdx.x;
  const int lane = tid & 31;
  const int wv   = tid >> 5;
  const int n0   = blockIdx.x * 64;
  const int mblk = blockIdx.y * 256;

  // --- stage B panel [64 x K] into LDS (16B chunks per lane) ---
  const int cpr    = K >> 3;          // 8-element (16B) chunks per row
  const int chunks = 64 * cpr;
  for (int idx = tid; idx < chunks; idx += 256) {
    const int r = idx / cpr, c = idx - r * cpr;
    const bf16* gsrc = B + (size_t)(n0 + r) * K + (c << 3);
    bf16* ldst = &Bs[r * BSTRIDE + (c << 3)];
#if ASYNC_STAGE
    // GLOBAL_LOAD_ASYNC_TO_LDS_B128: per-lane 16B memory -> LDS, ASYNCcnt.
    // VDST = LDS byte offset (low 32 bits of generic LDS address).
    const uint32_t ldsoff = (uint32_t)(uintptr_t)(void*)ldst;
    asm volatile("global_load_async_to_lds_b128 %0, %1, off"
                 :: "v"(ldsoff), "v"((uint64_t)(uintptr_t)gsrc)
                 : "memory");
#else
    *(uint4*)ldst = *(const uint4*)gsrc;
#endif
  }
#if ASYNC_STAGE
#if __has_builtin(__builtin_amdgcn_s_wait_asynccnt)
  __builtin_amdgcn_s_wait_asynccnt(0);
#else
  asm volatile("s_wait_asynccnt 0x0" ::: "memory");
#endif
#endif
  __syncthreads();

  v8f acc[2][4];
#pragma unroll
  for (int i = 0; i < 2; ++i)
#pragma unroll
    for (int t = 0; t < 4; ++t)
#pragma unroll
      for (int r = 0; r < 8; ++r) acc[i][t][r] = 0.f;

  // A fragment addressing per CDNA5 16-bit A layout:
  //   row = lane&15 ; K-base = (lane>>4)*8 ; chunks at +0 and +16 in K
  const int kb = (lane >> 4) << 3;
  const bf16* Ap[2];
#pragma unroll
  for (int i = 0; i < 2; ++i) {
    int arow = mblk + i * 128 + wv * 16 + (lane & 15);
    if (arow >= M) arow = M - 1;               // clamped; stores are guarded
    Ap[i] = A + (size_t)arow * K + kb;
  }

  for (int k0 = 0; k0 < K; k0 += 32) {
    __builtin_prefetch(Ap[0] + k0 + 64, 0, 1); // global_prefetch_b8
    __builtin_prefetch(Ap[1] + k0 + 64, 0, 1);
    FragAB a[2];
#pragma unroll
    for (int i = 0; i < 2; ++i) {
      a[i].h[0] = *(const v8bf*)(Ap[i] + k0);
      a[i].h[1] = *(const v8bf*)(Ap[i] + k0 + 16);
    }
#pragma unroll
    for (int t = 0; t < 4; ++t) {
      FragAB b;
      const bf16* bp = &Bs[(t * 16 + (lane & 15)) * BSTRIDE + k0 + kb];
      b.h[0] = *(const v8bf*)(bp);
      b.h[1] = *(const v8bf*)(bp + 16);
      acc[0][t] = __builtin_amdgcn_wmma_f32_16x16x32_bf16(
          false, a[0].v, false, b.v, (short)0, acc[0][t], false, false);
      acc[1][t] = __builtin_amdgcn_wmma_f32_16x16x32_bf16(
          false, a[1].v, false, b.v, (short)0, acc[1][t], false, false);
    }
  }

  // C layout: VGPR r holds M = r + 8*(lane>>4); N = lane&15 within tile
#pragma unroll
  for (int i = 0; i < 2; ++i) {
    const int mbase = mblk + i * 128 + wv * 16 + ((lane >> 4) << 3);
#pragma unroll
    for (int t = 0; t < 4; ++t) {
      const int n = n0 + t * 16 + (lane & 15);
      const float bv = bias ? bias[n] : 0.f;
#pragma unroll
      for (int r = 0; r < 8; ++r) {
        const int m = mbase + r;
        if (m < M) {
          float v = acc[i][t][r] + bv;
          if (relu) v = v > 0.f ? v : 0.f;
          const size_t off = (size_t)m * Nc + n;
          if (Cf) Cf[off] = v;
          if (Cb) Cb[off] = f2bf(v);
        }
      }
    }
  }
}

// ---------------------------------------------------------------------------
// Elementwise / graph utility kernels
// ---------------------------------------------------------------------------
__global__ void k_fill(float* p, float v, size_t n) {
  size_t i = (size_t)blockIdx.x * blockDim.x + threadIdx.x;
  if (i < n) p[i] = v;
}

__global__ void k_cast(const float* __restrict__ in, bf16* __restrict__ out, size_t n) {
  size_t i = (size_t)blockIdx.x * blockDim.x + threadIdx.x;
  if (i < n) out[i] = f2bf(in[i]);
}

__global__ void k_copy_cast(const float* __restrict__ in, float* __restrict__ of,
                            bf16* __restrict__ ob, size_t n) {
  size_t i = (size_t)blockIdx.x * blockDim.x + threadIdx.x;
  if (i < n) { float v = in[i]; of[i] = v; ob[i] = f2bf(v); }
}

__global__ void k_deg(const int* __restrict__ dst, float* __restrict__ deg, int ne) {
  int e = blockIdx.x * blockDim.x + threadIdx.x;
  if (e < ne) atomicAdd(&deg[dst[e]], 1.0f);
}

__global__ void k_rsqrt(float* __restrict__ d, int n) {
  int i = blockIdx.x * blockDim.x + threadIdx.x;
  if (i < n) { float v = d[i]; d[i] = v > 0.f ? rsqrtf(v) : 0.f; }
}

// agg[dst] += h[src] * dis[src]*dis[dst]   (64 threads/edge, float4 per thread)
__global__ void k_scatter(const float* __restrict__ h, const int* __restrict__ src,
                          const int* __restrict__ dst, const float* __restrict__ dis,
                          float* __restrict__ agg, int ne)
{
  long long gid = (long long)blockIdx.x * blockDim.x + threadIdx.x;
  if (gid >= (long long)ne * 64) return;
  const int e = (int)(gid >> 6);
  const int c = (int)(gid & 63);
  const int s = src[e], d = dst[e];
  const float nrm = dis[s] * dis[d];
  const float4 hv = *(const float4*)(h + (size_t)s * DIM + c * 4);
  float* o = agg + (size_t)d * DIM + c * 4;
  atomicAdd(o + 0, hv.x * nrm);
  atomicAdd(o + 1, hv.y * nrm);
  atomicAdd(o + 2, hv.z * nrm);
  atomicAdd(o + 3, hv.w * nrm);
}

// out = relu(agg + h*dis^2 + bias)  (self-loop term + bias + activation)
__global__ void k_gcn_final(const float* __restrict__ agg, const float* __restrict__ h,
                            const float* __restrict__ dis, const float* __restrict__ bias,
                            bf16* __restrict__ ob, float* __restrict__ of, size_t n)
{
  size_t i = (size_t)blockIdx.x * blockDim.x + threadIdx.x;
  if (i >= n) return;
  const int d    = (int)(i & (DIM - 1));
  const int node = (int)(i >> 8);
  const float di = dis[node];
  float v = agg[i] + h[i] * di * di + bias[d];
  v = v > 0.f ? v : 0.f;
  if (ob) ob[i] = f2bf(v);
  if (of) of[i] = v;
}

// ---------------------------------------------------------------------------
// Attention: S=8, 4 heads of 64. One thread per (node, s, head).
// qkv rows: token = n*8+t, cols [0,256)=Q [256,512)=K [512,768)=V.
// ---------------------------------------------------------------------------
__global__ void __launch_bounds__(256)
k_attn(const bf16* __restrict__ qkv, bf16* __restrict__ out, int nn)
{
  const int id = blockIdx.x * blockDim.x + threadIdx.x;
  if (id >= nn * SEQ * HEADS) return;
  const int h = id & (HEADS - 1);
  const int s = (id >> 2) & (SEQ - 1);
  const int n = id >> 5;

  float q[64];
  {
    const size_t rq = (size_t)(n * SEQ + s) * 768 + h * 64;
#pragma unroll
    for (int c = 0; c < 8; ++c) {
      v8bf t = *(const v8bf*)(qkv + rq + c * 8);
#pragma unroll
      for (int j = 0; j < 8; ++j) q[c * 8 + j] = bf2f(t[j]);
    }
  }
  float sc[SEQ]; float mx = -1e30f;
  for (int t = 0; t < SEQ; ++t) {
    const size_t rk = (size_t)(n * SEQ + t) * 768 + 256 + h * 64;
    float acc = 0.f;
#pragma unroll
    for (int c = 0; c < 8; ++c) {
      v8bf kv = *(const v8bf*)(qkv + rk + c * 8);
#pragma unroll
      for (int j = 0; j < 8; ++j) acc += q[c * 8 + j] * bf2f(kv[j]);
    }
    sc[t] = acc * 0.125f;                 // 1/sqrt(64)
    mx = fmaxf(mx, sc[t]);
  }
  float sum = 0.f;
#pragma unroll
  for (int t = 0; t < SEQ; ++t) { sc[t] = __expf(sc[t] - mx); sum += sc[t]; }
  const float inv = 1.f / sum;

  float o[64];
#pragma unroll
  for (int d = 0; d < 64; ++d) o[d] = 0.f;
  for (int t = 0; t < SEQ; ++t) {
    const size_t rv = (size_t)(n * SEQ + t) * 768 + 512 + h * 64;
    const float p = sc[t] * inv;
#pragma unroll
    for (int c = 0; c < 8; ++c) {
      v8bf vv = *(const v8bf*)(qkv + rv + c * 8);
#pragma unroll
      for (int j = 0; j < 8; ++j) o[c * 8 + j] += p * bf2f(vv[j]);
    }
  }
  const size_t ro = (size_t)(n * SEQ + s) * DIM + h * 64;
#pragma unroll
  for (int d = 0; d < 64; ++d) out[ro + d] = f2bf(o[d]);
}

// ---------------------------------------------------------------------------
// Fused residual + LayerNorm over D=256, one wave32 per token row.
// h = LN(h + dlt)*g + b ; writes f32 (residual stream) and bf16 (GEMM input).
// ---------------------------------------------------------------------------
__global__ void __launch_bounds__(256)
k_add_ln(float* __restrict__ h, const float* __restrict__ dlt,
         const float* __restrict__ g, const float* __restrict__ b,
         bf16* __restrict__ hb, int rows)
{
  const int widx = (int)(((size_t)blockIdx.x * blockDim.x + threadIdx.x) >> 5);
  const int lane = threadIdx.x & 31;
  if (widx >= rows) return;
  const size_t base = (size_t)widx * DIM;

  float v[8]; float s = 0.f, s2 = 0.f;
#pragma unroll
  for (int i = 0; i < 8; ++i) {
    const int d = lane + i * 32;
    v[i] = h[base + d] + dlt[base + d];
    s += v[i]; s2 += v[i] * v[i];
  }
#pragma unroll
  for (int off = 16; off > 0; off >>= 1) {
    s  += __shfl_xor(s,  off);
    s2 += __shfl_xor(s2, off);
  }
  const float mu   = s * (1.f / DIM);
  const float var  = s2 * (1.f / DIM) - mu * mu;
  const float rstd = rsqrtf(var + 1e-5f);
#pragma unroll
  for (int i = 0; i < 8; ++i) {
    const int d = lane + i * 32;
    const float y = (v[i] - mu) * rstd * g[d] + b[d];
    h[base + d]  = y;
    hb[base + d] = f2bf(y);
  }
}

// ---------------------------------------------------------------------------
// Final: sem = mean_s(h), combined = [g | sem] (512), out = combined @ fc_w^T + fc_b
// One wave32 per node; K=512, N=10 -> skinny, VALU + shfl reduce.
// ---------------------------------------------------------------------------
__global__ void __launch_bounds__(256)
k_final(const float* __restrict__ g, const float* __restrict__ h,
        const float* __restrict__ fcw, const float* __restrict__ fcb,
        float* __restrict__ out, int nn)
{
  const int widx = (int)(((size_t)blockIdx.x * blockDim.x + threadIdx.x) >> 5);
  const int lane = threadIdx.x & 31;
  if (widx >= nn) return;

  float comb[16];
#pragma unroll
  for (int i = 0; i < 8; ++i) comb[i] = g[(size_t)widx * DIM + lane + i * 32];
#pragma unroll
  for (int i = 0; i < 8; ++i) {
    float s = 0.f;
    for (int t = 0; t < SEQ; ++t)
      s += h[(size_t)(widx * SEQ + t) * DIM + lane + i * 32];
    comb[8 + i] = s * (1.f / SEQ);
  }
  for (int c = 0; c < NCLS; ++c) {
    float acc = 0.f;
#pragma unroll
    for (int i = 0; i < 16; ++i)
      acc += comb[i] * fcw[(size_t)c * 512 + i * 32 + lane];
#pragma unroll
    for (int off = 16; off > 0; off >>= 1) acc += __shfl_xor(acc, off);
    if (lane == 0) out[(size_t)widx * NCLS + c] = acc + fcb[c];
  }
}

// ---------------------------------------------------------------------------
// Host launcher
// ---------------------------------------------------------------------------
extern "C" void kernel_launch(void* const* d_in, const int* in_sizes, int n_in,
                              void* d_out, int out_size, void* d_ws, size_t ws_size,
                              hipStream_t stream)
{
  (void)in_sizes; (void)n_in; (void)out_size; (void)ws_size;

  const float* x        = (const float*)d_in[0];
  const int*   ei       = (const int*)  d_in[1];
  const float* semantic = (const float*)d_in[2];
  const float* g1w = (const float*)d_in[3];
  const float* g1b = (const float*)d_in[4];
  const float* g2w = (const float*)d_in[5];
  const float* g2b = (const float*)d_in[6];
  const float* qkvw = (const float*)d_in[7];
  const float* qkvb = (const float*)d_in[8];
  const float* outw = (const float*)d_in[9];
  const float* outb = (const float*)d_in[10];
  const float* ln1g = (const float*)d_in[11];
  const float* ln1b = (const float*)d_in[12];
  const float* f1w  = (const float*)d_in[13];
  const float* f1b  = (const float*)d_in[14];
  const float* f2w  = (const float*)d_in[15];
  const float* f2b  = (const float*)d_in[16];
  const float* ln2g = (const float*)d_in[17];
  const float* ln2b = (const float*)d_in[18];
  const float* fcw  = (const float*)d_in[19];
  const float* fcb  = (const float*)d_in[20];
  float* out = (float*)d_out;

  const int* srcI = ei;
  const int* dstI = ei + N_EDGES;

  // bump allocator over d_ws
  char* wp = (char*)d_ws;
  auto alloc = [&](size_t bytes) -> void* {
    void* r = (void*)wp;
    wp += (bytes + 255) & ~(size_t)255;
    return r;
  };
  const size_t T = (size_t)N_NODES * SEQ;       // 400000 tokens

  float* deg   = (float*)alloc((size_t)N_NODES * 4);
  bf16*  xbf   = (bf16*) alloc((size_t)N_NODES * DIM * 2);
  float* hg    = (float*)alloc((size_t)N_NODES * DIM * 4);
  float* agg   = (float*)alloc((size_t)N_NODES * DIM * 4);
  bf16*  gbf   = (bf16*) alloc((size_t)N_NODES * DIM * 2);
  float* gfin  = (float*)alloc((size_t)N_NODES * DIM * 4);
  bf16*  g1wb  = (bf16*) alloc((size_t)HIDN * DIM * 2);
  bf16*  g2wb  = (bf16*) alloc((size_t)HIDN * DIM * 2);
  bf16*  qkvwb = (bf16*) alloc((size_t)LAYERS * 3 * DIM * DIM * 2);
  bf16*  outwb = (bf16*) alloc((size_t)LAYERS * DIM * DIM * 2);
  bf16*  f1wb  = (bf16*) alloc((size_t)LAYERS * HIDN * DIM * 2);
  bf16*  f2wb  = (bf16*) alloc((size_t)LAYERS * DIM * HIDN * 2);
  float* hf    = (float*)alloc(T * DIM * 4);
  bf16*  hb    = (bf16*) alloc(T * DIM * 2);
  bf16*  qkv_s = (bf16*) alloc(T * 3 * DIM * 2);
  bf16*  ao_s  = (bf16*) alloc(T * DIM * 2);
  float* tf    = (float*)alloc(T * DIM * 4);
  bf16*  f1_s  = (bf16*) alloc(T * DIM * 2);

  auto cdiv = [](size_t a, size_t b) { return (unsigned)((a + b - 1) / b); };
  const dim3 blk(256);

  // ---- weight / input casts to bf16 ----
  k_cast<<<cdiv((size_t)N_NODES * DIM, 256), blk, 0, stream>>>(x, xbf, (size_t)N_NODES * DIM);
  k_cast<<<cdiv((size_t)HIDN * DIM, 256), blk, 0, stream>>>(g1w, g1wb, (size_t)HIDN * DIM);
  k_cast<<<cdiv((size_t)HIDN * DIM, 256), blk, 0, stream>>>(g2w, g2wb, (size_t)HIDN * DIM);
  k_cast<<<cdiv((size_t)LAYERS * 3 * DIM * DIM, 256), blk, 0, stream>>>(qkvw, qkvwb, (size_t)LAYERS * 3 * DIM * DIM);
  k_cast<<<cdiv((size_t)LAYERS * DIM * DIM, 256), blk, 0, stream>>>(outw, outwb, (size_t)LAYERS * DIM * DIM);
  k_cast<<<cdiv((size_t)LAYERS * HIDN * DIM, 256), blk, 0, stream>>>(f1w, f1wb, (size_t)LAYERS * HIDN * DIM);
  k_cast<<<cdiv((size_t)LAYERS * DIM * HIDN, 256), blk, 0, stream>>>(f2w, f2wb, (size_t)LAYERS * DIM * HIDN);

  // ---- degrees (self-loop => init 1.0), then dis = rsqrt(deg) in place ----
  k_fill<<<cdiv((size_t)N_NODES, 256), blk, 0, stream>>>(deg, 1.0f, (size_t)N_NODES);
  k_deg<<<cdiv((size_t)N_EDGES, 256), blk, 0, stream>>>(dstI, deg, N_EDGES);
  k_rsqrt<<<cdiv((size_t)N_NODES, 256), blk, 0, stream>>>(deg, N_NODES);

  // ---- GCN layer 1 ----
  k_gemm<<<dim3(DIM / 64, cdiv(N_NODES, 256)), blk, 0, stream>>>(
      xbf, g1wb, nullptr, hg, nullptr, N_NODES, DIM, DIM, 0);
  k_fill<<<cdiv((size_t)N_NODES * DIM, 256), blk, 0, stream>>>(agg, 0.f, (size_t)N_NODES * DIM);
  k_scatter<<<cdiv((size_t)N_EDGES * 64, 256), blk, 0, stream>>>(hg, srcI, dstI, deg, agg, N_EDGES);
  k_gcn_final<<<cdiv((size_t)N_NODES * DIM, 256), blk, 0, stream>>>(
      agg, hg, deg, g1b, gbf, nullptr, (size_t)N_NODES * DIM);

  // ---- GCN layer 2 ----
  k_gemm<<<dim3(DIM / 64, cdiv(N_NODES, 256)), blk, 0, stream>>>(
      gbf, g2wb, nullptr, hg, nullptr, N_NODES, DIM, DIM, 0);
  k_fill<<<cdiv((size_t)N_NODES * DIM, 256), blk, 0, stream>>>(agg, 0.f, (size_t)N_NODES * DIM);
  k_scatter<<<cdiv((size_t)N_EDGES * 64, 256), blk, 0, stream>>>(hg, srcI, dstI, deg, agg, N_EDGES);
  k_gcn_final<<<cdiv((size_t)N_NODES * DIM, 256), blk, 0, stream>>>(
      agg, hg, deg, g2b, nullptr, gfin, (size_t)N_NODES * DIM);

  // ---- transformer: init residual stream from semantic ----
  k_copy_cast<<<cdiv(T * DIM, 256), blk, 0, stream>>>(semantic, hf, hb, T * DIM);

  for (int l = 0; l < LAYERS; ++l) {
    const bf16* qw = qkvwb + (size_t)l * 3 * DIM * DIM;
    // QKV projection: [T,256] x [768,256]^T -> bf16 [T,768]
    k_gemm<<<dim3((3 * DIM) / 64, cdiv(T, 256)), blk, 0, stream>>>(
        hb, qw, qkvb + (size_t)l * 3 * DIM, nullptr, qkv_s, (int)T, 3 * DIM, DIM, 0);
    // attention (S=8, 4 heads)
    k_attn<<<cdiv((size_t)N_NODES * SEQ * HEADS, 256), blk, 0, stream>>>(qkv_s, ao_s, N_NODES);
    // output projection -> f32 [T,256]
    k_gemm<<<dim3(DIM / 64, cdiv(T, 256)), blk, 0, stream>>>(
        ao_s, outwb + (size_t)l * DIM * DIM, outb + (size_t)l * DIM, tf, nullptr, (int)T, DIM, DIM, 0);
    // h = LN(h + o)
    k_add_ln<<<cdiv(T * 32, 256), blk, 0, stream>>>(
        hf, tf, ln1g + (size_t)l * DIM, ln1b + (size_t)l * DIM, hb, (int)T);
    // FFN: relu(h @ W1^T + b1) -> bf16 [T,256]
    k_gemm<<<dim3(HIDN / 64, cdiv(T, 256)), blk, 0, stream>>>(
        hb, f1wb + (size_t)l * HIDN * DIM, f1b + (size_t)l * HIDN, nullptr, f1_s, (int)T, HIDN, DIM, 1);
    // FFN out -> f32 [T,256]
    k_gemm<<<dim3(DIM / 64, cdiv(T, 256)), blk, 0, stream>>>(
        f1_s, f2wb + (size_t)l * DIM * HIDN, f2b + (size_t)l * DIM, tf, nullptr, (int)T, DIM, HIDN, 0);
    // h = LN(h + ff)
    k_add_ln<<<cdiv(T * 32, 256), blk, 0, stream>>>(
        hf, tf, ln2g + (size_t)l * DIM, ln2b + (size_t)l * DIM, hb, (int)T);
  }

  // ---- mean over S, concat with GCN output, final FC ----
  k_final<<<cdiv((size_t)N_NODES * 32, 256), blk, 0, stream>>>(gfin, hf, fcw, fcb, out, N_NODES);
}
